// GATDecoder_20882130993769
// MI455X (gfx1250) — compile-verified, matrix operand order checked by reading, not run on hardware
//
#include <hip/hip_runtime.h>
#include <hip/hip_bf16.h>

// ---------------------------------------------------------------------------
// Problem constants (match reference)
// ---------------------------------------------------------------------------
#define B_      64
#define N_ORG_  224
#define N_PAD_  32
#define M_      256
#define S_      128
#define NTOT_   (B_ * M_)        // 16384
#define D_      256
#define H_      8
#define CH_     32               // D/H
#define DEG_    16
#define E_ORG_  (B_ * N_ORG_ * DEG_)   // 229376
#define E_PAD_  (B_ * N_PAD_ * DEG_)   // 32768
#define E_      (E_ORG_ + E_PAD_)      // 262144
#define NEG_SLOPE_ 0.2f

// ---------------------------------------------------------------------------
// WMMA types (CDNA5, wave32): v_wmma_f32_16x16x32_bf16
// ---------------------------------------------------------------------------
typedef __attribute__((ext_vector_type(16))) __bf16 v16bf;
typedef __attribute__((ext_vector_type(8)))  float  v8f;

union FragBF { v16bf v; unsigned u[8]; };

__device__ __forceinline__ unsigned short f2bf(float f) {
  unsigned u = __float_as_uint(f);
  return (unsigned short)((u + 0x7fffu + ((u >> 16) & 1u)) >> 16); // RNE
}
__device__ __forceinline__ unsigned pk2(float a, float b) {
  return (unsigned)f2bf(a) | ((unsigned)f2bf(b) << 16);
}
__device__ __forceinline__ float reluf(float v) { return v > 0.f ? v : 0.f; }

// ---------------------------------------------------------------------------
// Tiled bf16 WMMA GEMM:  C[M,N] = act( A[M,K] @ W[K,N] + bias (+ resid) )
// Block: 256 threads (8 wave32 waves), tile 128(M) x 64(N), K step 32.
// Each wave: 32x32 tile = 2x2 WMMA 16x16x32 fragments.
// AMODE 0: A is plain row-major [M,K]
// AMODE 1: A row m = relu(concat(node[idx0[m]], node[idx1[m]])), K=512
// AMODE 2: A row m = concat(edge[m], node[idx0[m]], node[idx1[m]]), K=768
// MALIGN : M is a multiple of 128 (drop all row-bound checks)
// N is always 256 and grids are exact in N, so no N checks anywhere.
// ---------------------------------------------------------------------------
#define TILE_M 128
#define TILE_N 64
#define TILE_K 32
#define LDSB   (TILE_K + 2)   // ushort stride; even => 4B-aligned pair loads

template<int AMODE>
__device__ __forceinline__ const float* rowPtrA(const float* A, const float* Aedge,
                                                const float* Anode,
                                                const int* idx0, const int* idx1,
                                                int m, int k0, int K) {
  if (AMODE == 0) {
    return A + (size_t)m * K + k0;
  } else if (AMODE == 1) {
    int r = (k0 < 256) ? idx0[m] : idx1[m];
    return Anode + (size_t)r * D_ + (k0 & 255);
  } else {
    if (k0 < 256) return Aedge + (size_t)m * D_ + k0;
    int r = (k0 < 512) ? idx0[m] : idx1[m];
    return Anode + (size_t)r * D_ + (k0 & 255);
  }
}

template<int AMODE, bool RELUO, bool RES, bool MALIGN>
__global__ __launch_bounds__(256) void gemm_wmma(
    const float* __restrict__ A, const float* __restrict__ Aedge,
    const float* __restrict__ Anode, const int* __restrict__ idx0,
    const int* __restrict__ idx1, const float* __restrict__ W,
    const float* __restrict__ bias, const float* __restrict__ resid,
    float* __restrict__ C, int M, int N, int K) {
  __shared__ unsigned short As[TILE_M][LDSB];
  __shared__ unsigned short Bs[TILE_N][LDSB];

  const int tid  = threadIdx.x;
  const int lane = tid & 31;
  const int wave = tid >> 5;
  const int wr   = wave >> 1;      // 0..3  (M sub-tile)
  const int wc   = wave & 1;       // 0..1  (N sub-tile)
  const int m0   = blockIdx.y * TILE_M;
  const int n0   = blockIdx.x * TILE_N;

  v8f acc[2][2];
#pragma unroll
  for (int i = 0; i < 2; ++i)
#pragma unroll
    for (int j = 0; j < 2; ++j)
#pragma unroll
      for (int r = 0; r < 8; ++r) acc[i][j][r] = 0.f;

  const int lr  = tid >> 3;        // 0..31 (A row group; +32*i covers 128)
  const int lc4 = (tid & 7) * 4;   // A col base (4 floats / thread / row)
  const int bkr = tid >> 3;        // 0..31 (B k row)
  const int bnb = (tid & 7) * 8;   // B col base (8 floats / thread)

  for (int k0 = 0; k0 < K; k0 += TILE_K) {
    // ---- stage A tile (128 x 32) into LDS as packed bf16 pairs ----
#pragma unroll
    for (int i = 0; i < 4; ++i) {
      int r  = lr + 32 * i;
      int gm = m0 + r;
      unsigned* dst = (unsigned*)&As[r][lc4];
      if (MALIGN || gm < M) {
        const float* ap =
            rowPtrA<AMODE>(A, Aedge, Anode, idx0, idx1, gm, k0, K) + lc4;
        if (AMODE == 0 && k0 + TILE_K < K)
          __builtin_prefetch(ap + TILE_K, 0, 1);
        float4 v = *(const float4*)ap;              // global_load_b128
        if (AMODE == 1) {
          v.x = reluf(v.x); v.y = reluf(v.y); v.z = reluf(v.z); v.w = reluf(v.w);
        }
        dst[0] = pk2(v.x, v.y);                     // ds_store_b32 x2
        dst[1] = pk2(v.z, v.w);
      } else {
        dst[0] = 0u; dst[1] = 0u;
      }
    }
    // ---- stage B tile (32 x 64) into LDS transposed: Bs[n][k] ----
    {
      const float* wp = W + (size_t)(k0 + bkr) * N + n0 + bnb;
      float4 w0 = *(const float4*)wp;               // global_load_b128 x2
      float4 w1 = *(const float4*)(wp + 4);
      Bs[bnb + 0][bkr] = f2bf(w0.x);
      Bs[bnb + 1][bkr] = f2bf(w0.y);
      Bs[bnb + 2][bkr] = f2bf(w0.z);
      Bs[bnb + 3][bkr] = f2bf(w0.w);
      Bs[bnb + 4][bkr] = f2bf(w1.x);
      Bs[bnb + 5][bkr] = f2bf(w1.y);
      Bs[bnb + 6][bkr] = f2bf(w1.z);
      Bs[bnb + 7][bkr] = f2bf(w1.w);
    }
    __syncthreads();

    // ---- build fragments per ISA 7.12.2 16-bit layout ----
    const int g8 = (lane >> 4) * 8;   // lane-group K base (0 or 8)
    const int mr = lane & 15;
    FragBF afr[2], bfr[2];
#pragma unroll
    for (int s = 0; s < 2; ++s) {
      int ar = wr * 32 + s * 16 + mr;
      int bc = wc * 32 + s * 16 + mr;
#pragma unroll
      for (int j = 0; j < 4; ++j) {
        afr[s].u[j]     = *(const unsigned*)&As[ar][g8 + 2 * j];
        afr[s].u[4 + j] = *(const unsigned*)&As[ar][16 + g8 + 2 * j];
        bfr[s].u[j]     = *(const unsigned*)&Bs[bc][g8 + 2 * j];
        bfr[s].u[4 + j] = *(const unsigned*)&Bs[bc][16 + g8 + 2 * j];
      }
    }
#pragma unroll
    for (int si = 0; si < 2; ++si)
#pragma unroll
      for (int sj = 0; sj < 2; ++sj)
        acc[si][sj] = __builtin_amdgcn_wmma_f32_16x16x32_bf16(
            false, afr[si].v, false, bfr[sj].v, (short)0, acc[si][sj],
            false, false);
    __syncthreads();
  }

  // ---- epilogue: C/D layout (VGPR r -> M=r / r+8 by lane group, N=lane&15) --
#pragma unroll
  for (int si = 0; si < 2; ++si) {
#pragma unroll
    for (int sj = 0; sj < 2; ++sj) {
      int nn = n0 + wc * 32 + sj * 16 + (lane & 15);
      float bv = bias ? bias[nn] : 0.f;
#pragma unroll
      for (int r = 0; r < 8; ++r) {
        int mm = m0 + wr * 32 + si * 16 + r + ((lane >> 4) ? 8 : 0);
        if (MALIGN || mm < M) {
          float v = acc[si][sj][r] + bv;
          if (RES)   v += resid[(size_t)mm * N + nn];
          if (RELUO) v = reluf(v);
          C[(size_t)mm * N + nn] = v;
        }
      }
    }
  }
}

// ---------------------------------------------------------------------------
// Cross-attention: per (b, h) block; online-softmax, K/V staged in LDS.
// q row layout: [rows, D] with head h at cols h*32..h*32+31.
// ---------------------------------------------------------------------------
#define KVP 36  // float stride; multiple of 4 keeps float4 alignment in LDS

__global__ __launch_bounds__(256) void attn_kernel(
    const float* __restrict__ q, const float* __restrict__ k,
    const float* __restrict__ v, float* __restrict__ out,
    int Lq, int Lk, float scale, int qShared) {
  const int b = blockIdx.x, h = blockIdx.y;
  __shared__ float Ks[S_][KVP];
  __shared__ float Vs[S_][KVP];
  const int tid = threadIdx.x;
  const int lr = tid >> 3, lc = (tid & 7) * 4;
#pragma unroll
  for (int i = 0; i < 4; ++i) {
    int r = lr + 32 * i;
    if (r < Lk) {
      const float* kp = k + ((size_t)(b * Lk + r)) * D_ + h * CH_ + lc;
      const float* vp = v + ((size_t)(b * Lk + r)) * D_ + h * CH_ + lc;
      *(float4*)&Ks[r][lc] = *(const float4*)kp;
      *(float4*)&Vs[r][lc] = *(const float4*)vp;
    }
  }
  __syncthreads();
  const int row = tid;
  if (row >= Lq) return;
  const float* qp = q + (qShared ? (size_t)row * D_
                                 : ((size_t)(b * Lq + row)) * D_) + h * CH_;
  float qr[CH_], accv[CH_];
#pragma unroll
  for (int c = 0; c < CH_; ++c) { qr[c] = qp[c]; accv[c] = 0.f; }
  float m = -1e30f, l = 0.f;
  for (int j = 0; j < Lk; ++j) {
    float s = 0.f;
#pragma unroll
    for (int c = 0; c < CH_; ++c) s += qr[c] * Ks[j][c];
    s *= scale;
    float nm = fmaxf(m, s);
    float corr = __expf(m - nm);
    float p = __expf(s - nm);
    l = l * corr + p;
#pragma unroll
    for (int c = 0; c < CH_; ++c) accv[c] = accv[c] * corr + p * Vs[j][c];
    m = nm;
  }
  float inv = 1.f / l;
  float* op = out + ((size_t)(b * Lq + row)) * D_ + h * CH_;
#pragma unroll
  for (int c = 0; c < CH_; ++c) op[c] = accv[c] * inv;
}

// ---------------------------------------------------------------------------
// LayerNorm (+ReLU, optional residual add AFTER relu): one wave per row (D=256)
// out = relu(LN(x)*g + b) [+ res]
// ---------------------------------------------------------------------------
template<bool ADDRES>
__global__ __launch_bounds__(256) void ln_relu_kernel(
    const float* __restrict__ x, const float* __restrict__ g,
    const float* __restrict__ bta, const float* __restrict__ res,
    float* __restrict__ out, int rows) {
  const int wave = threadIdx.x >> 5, lane = threadIdx.x & 31;
  const long long row = (long long)blockIdx.x * 8 + wave;
  if (row >= rows) return;
  const float* xp = x + row * D_;
  float vals[8];
  float s = 0.f;
#pragma unroll
  for (int i = 0; i < 8; ++i) { vals[i] = xp[lane + 32 * i]; s += vals[i]; }
#pragma unroll
  for (int off = 16; off > 0; off >>= 1) s += __shfl_xor(s, off, 32);
  const float mean = s * (1.f / 256.f);
  float vs = 0.f;
#pragma unroll
  for (int i = 0; i < 8; ++i) { float d = vals[i] - mean; vs += d * d; }
#pragma unroll
  for (int off = 16; off > 0; off >>= 1) vs += __shfl_xor(vs, off, 32);
  const float rstd = rsqrtf(vs * (1.f / 256.f) + 1e-5f);
  float* op = out + row * D_;
  const float* rp = res + row * D_;
#pragma unroll
  for (int i = 0; i < 8; ++i) {
    int c = lane + 32 * i;
    float y = (vals[i] - mean) * rstd * g[c] + bta[c];
    y = y > 0.f ? y : 0.f;
    if (ADDRES) y += rp[c];
    op[c] = y;
  }
}

// ---------------------------------------------------------------------------
// Small elementwise / gather / scatter kernels
// ---------------------------------------------------------------------------
__global__ void k_embed_org(const int* __restrict__ x,
                            const float* __restrict__ tab,
                            float* __restrict__ node) {
  long long t = (long long)blockIdx.x * blockDim.x + threadIdx.x;
  if (t >= (long long)B_ * N_ORG_ * D_) return;
  int c = (int)(t & 255);
  long long r = t >> 8;
  int b = (int)(r / N_ORG_), i = (int)(r % N_ORG_);
  node[((size_t)(b * M_ + i)) * D_ + c] = tab[(size_t)x[r] * D_ + c];
}

__global__ void k_embed_bond(const int* __restrict__ ea,
                             const float* __restrict__ tab,
                             float* __restrict__ edge) {
  long long t = (long long)blockIdx.x * blockDim.x + threadIdx.x;
  if (t >= (long long)E_ORG_ * D_) return;
  int c = (int)(t & 255);
  long long e = t >> 8;
  edge[(size_t)e * D_ + c] = tab[(size_t)ea[e] * D_ + c];
}

__global__ void k_copy_pad(const float* __restrict__ src, float* __restrict__ node) {
  long long t = (long long)blockIdx.x * blockDim.x + threadIdx.x;
  if (t >= (long long)B_ * N_PAD_ * D_) return;
  int c = (int)(t & 255);
  long long r = t >> 8;
  int b = (int)(r / N_PAD_), i = (int)(r % N_PAD_);
  node[((size_t)(b * M_ + N_ORG_ + i)) * D_ + c] = src[(size_t)r * D_ + c];
}

__global__ void k_fill(float* __restrict__ p, float v, long long n) {
  long long t = (long long)blockIdx.x * blockDim.x + threadIdx.x;
  if (t < n) p[t] = v;
}

__global__ void k_fill_rows_bias(float* __restrict__ p, const float* __restrict__ b,
                                 long long rows) {
  long long t = (long long)blockIdx.x * blockDim.x + threadIdx.x;
  if (t >= rows * D_) return;
  p[t] = b[t & 255];
}

// out[r*H+h] = sum_c X[r*D + h*32 + c] * att[h*32+c]
__global__ void k_head_reduce(const float* __restrict__ X,
                              const float* __restrict__ att,
                              float* __restrict__ out, long long rows) {
  long long t = (long long)blockIdx.x * blockDim.x + threadIdx.x;
  if (t >= rows * H_) return;
  long long r = t >> 3;
  int h = (int)(t & 7);
  const float* xp = X + r * D_ + h * CH_;
  const float* ap = att + h * CH_;
  float s = 0.f;
#pragma unroll
  for (int c = 0; c < CH_; ++c) s += xp[c] * ap[c];
  out[t] = s;
}

__device__ __forceinline__ void atomicMaxF(float* a, float v) {
  float cur = *a;
  while (v > cur) {
    unsigned oldu = atomicCAS((unsigned*)a, __float_as_uint(cur), __float_as_uint(v));
    float prev = __uint_as_float(oldu);
    if (prev == cur) break;
    cur = prev;
  }
}

__global__ void k_logits_max(const float* __restrict__ asrc,
                             const float* __restrict__ adst,
                             const float* __restrict__ ae,
                             const int* __restrict__ src,
                             const int* __restrict__ dst,
                             float* __restrict__ lg, float* __restrict__ mx) {
  long long t = (long long)blockIdx.x * blockDim.x + threadIdx.x;
  if (t >= (long long)E_ * H_) return;
  long long e = t >> 3;
  int h = (int)(t & 7);
  float x = asrc[(size_t)src[e] * H_ + h] + adst[(size_t)dst[e] * H_ + h] + ae[t];
  x = x > 0.f ? x : NEG_SLOPE_ * x;
  lg[t] = x;
  atomicMaxF(&mx[(size_t)dst[e] * H_ + h], x);
}

__global__ void k_exp_den(float* __restrict__ lg, const float* __restrict__ mx,
                          const int* __restrict__ dst, float* __restrict__ den) {
  long long t = (long long)blockIdx.x * blockDim.x + threadIdx.x;
  if (t >= (long long)E_ * H_) return;
  long long e = t >> 3;
  int h = (int)(t & 7);
  float ex = __expf(lg[t] - mx[(size_t)dst[e] * H_ + h]);
  lg[t] = ex;
  atomicAdd(&den[(size_t)dst[e] * H_ + h], ex);
}

__global__ void k_scatter_msg(const float* __restrict__ ex,
                              const float* __restrict__ den,
                              const int* __restrict__ src,
                              const int* __restrict__ dst,
                              const float* __restrict__ Hbuf,
                              float* __restrict__ outb) {
  long long t = (long long)blockIdx.x * blockDim.x + threadIdx.x;
  if (t >= (long long)E_ * H_) return;
  long long e = t >> 3;
  int h = (int)(t & 7);
  int d = dst[e], s = src[e];
  float alpha = ex[t] / (den[(size_t)d * H_ + h] + 1e-16f);
  const float* hp = Hbuf + (size_t)s * D_ + h * CH_;
  float* op = outb + (size_t)d * D_ + h * CH_;
#pragma unroll
  for (int c = 0; c < CH_; ++c) atomicAdd(&op[c], hp[c] * alpha);
}

// ---------------------------------------------------------------------------
// Host orchestration
// ---------------------------------------------------------------------------
static inline int cdiv_ll(long long n, int d) { return (int)((n + d - 1) / d); }

extern "C" void kernel_launch(void* const* d_in, const int* in_sizes, int n_in,
                              void* d_out, int out_size, void* d_ws, size_t ws_size,
                              hipStream_t stream) {
  (void)in_sizes; (void)n_in; (void)out_size; (void)ws_size;
  // ---- data inputs ----
  const int*   x        = (const int*)d_in[0];
  const int*   edgeattr = (const int*)d_in[1];
  const int*   ei       = (const int*)d_in[2];         // (2, E) flattened
  const float* memory   = (const float*)d_in[3];
  const int* srcI = ei;        // edge_index[0]
  const int* dstI = ei + E_;   // edge_index[1]

  // ---- param leaves: jax tree-flatten (sorted-key) order after 4 data inputs:
  // 4 Qemb, 5 W_ext, 6 atom_emb, 7 b_ext, 8 bond_emb,
  // 9..16 init_attn {Wk,Wo,Wq,Wv,bk,bo,bq,bv},
  // 17+24l+: attn{Wk,Wo,Wq,Wv,bk,bo,bq,bv}(0-7),
  //          conv{W,W_edge,att_dst,att_edge,att_src,bias}(8-13),
  //          eu{W1,W2,b1,b2,ln_b,ln_g}(14-19),
  //          ln1_b(20), ln1_g(21), ln2_b(22), ln2_g(23)
  auto P = [&](int i) { return (const float*)d_in[i]; };

  // ---- outputs live in d_out: node then edge ----
  float* node = (float*)d_out;
  float* edge = node + (size_t)NTOT_ * D_;

  // ---- workspace carve-up (floats) ----
  float* w = (float*)d_ws;
  size_t off = 0;
  auto alloc = [&](size_t n) { float* p = w + off; off += n; return p; };
  float* Hbuf    = alloc((size_t)NTOT_ * D_);
  float* Ebuf    = alloc((size_t)E_ * D_);
  float* qbuf    = alloc((size_t)NTOT_ * D_);
  float* kbuf    = alloc((size_t)B_ * S_ * D_);
  float* vbuf    = alloc((size_t)B_ * S_ * D_);
  float* attnbuf = alloc((size_t)NTOT_ * D_);
  float* crbuf   = alloc((size_t)NTOT_ * D_);
  float* convbuf = alloc((size_t)NTOT_ * D_);
  float* asrc    = alloc((size_t)NTOT_ * H_);
  float* adst    = alloc((size_t)NTOT_ * H_);
  float* aeb     = alloc((size_t)E_ * H_);
  float* lgbuf   = alloc((size_t)E_ * H_);
  float* mxb     = alloc((size_t)NTOT_ * H_);
  float* denb    = alloc((size_t)NTOT_ * H_);

  const float scale = 0.17677669529663687f; // 1/sqrt(32)
  auto ggrid = [&](int Mr) { return dim3(D_ / TILE_N, (Mr + TILE_M - 1) / TILE_M); };
  dim3 blk(256);

  // ============================ init ============================
  k_embed_org<<<cdiv_ll((long long)B_ * N_ORG_ * D_, 256), blk, 0, stream>>>(x, P(6), node);

  // q0 = Qemb @ Wq + bq  (shared across batch; M=32 -> unaligned variant)
  gemm_wmma<0,false,false,false><<<ggrid(N_PAD_), blk, 0, stream>>>(
      P(4), nullptr, nullptr, nullptr, nullptr, P(11), P(15), nullptr, qbuf,
      N_PAD_, D_, D_);
  gemm_wmma<0,false,false,true><<<ggrid(B_ * S_), blk, 0, stream>>>(
      memory, nullptr, nullptr, nullptr, nullptr, P(9), P(13), nullptr, kbuf,
      B_ * S_, D_, D_);
  gemm_wmma<0,false,false,true><<<ggrid(B_ * S_), blk, 0, stream>>>(
      memory, nullptr, nullptr, nullptr, nullptr, P(12), P(16), nullptr, vbuf,
      B_ * S_, D_, D_);
  attn_kernel<<<dim3(B_, H_), blk, 0, stream>>>(qbuf, kbuf, vbuf, attnbuf,
                                                N_PAD_, S_, scale, 1);
  gemm_wmma<0,false,false,true><<<ggrid(B_ * N_PAD_), blk, 0, stream>>>(
      attnbuf, nullptr, nullptr, nullptr, nullptr, P(10), P(14), nullptr, crbuf,
      B_ * N_PAD_, D_, D_);
  k_copy_pad<<<cdiv_ll((long long)B_ * N_PAD_ * D_, 256), blk, 0, stream>>>(crbuf, node);

  // edge features: bond embeddings + pe = relu(concat(node[row],node[col])) @ W_ext + b_ext
  k_embed_bond<<<cdiv_ll((long long)E_ORG_ * D_, 256), blk, 0, stream>>>(edgeattr, P(8), edge);
  gemm_wmma<1,false,false,true><<<ggrid(E_PAD_), blk, 0, stream>>>(
      nullptr, nullptr, node, srcI + E_ORG_, dstI + E_ORG_, P(5), P(7), nullptr,
      edge + (size_t)E_ORG_ * D_, E_PAD_, D_, 2 * D_);

  // ============================ layers ============================
  for (int l = 0; l < 4; ++l) {
    const int base = 17 + 24 * l;

    // ---- gat_conv ----
    gemm_wmma<0,false,false,true><<<ggrid(NTOT_), blk, 0, stream>>>(
        node, nullptr, nullptr, nullptr, nullptr, P(base + 8), nullptr, nullptr,
        Hbuf, NTOT_, D_, D_);
    gemm_wmma<0,false,false,true><<<ggrid(E_), blk, 0, stream>>>(
        edge, nullptr, nullptr, nullptr, nullptr, P(base + 9), nullptr, nullptr,
        Ebuf, E_, D_, D_);
    k_head_reduce<<<cdiv_ll((long long)NTOT_ * H_, 256), blk, 0, stream>>>(
        Hbuf, P(base + 12), asrc, NTOT_);
    k_head_reduce<<<cdiv_ll((long long)NTOT_ * H_, 256), blk, 0, stream>>>(
        Hbuf, P(base + 10), adst, NTOT_);
    k_head_reduce<<<cdiv_ll((long long)E_ * H_, 256), blk, 0, stream>>>(
        Ebuf, P(base + 11), aeb, E_);
    k_fill<<<cdiv_ll((long long)NTOT_ * H_, 256), blk, 0, stream>>>(mxb, -1e30f,
        (long long)NTOT_ * H_);
    k_fill<<<cdiv_ll((long long)NTOT_ * H_, 256), blk, 0, stream>>>(denb, 0.f,
        (long long)NTOT_ * H_);
    k_fill_rows_bias<<<cdiv_ll((long long)NTOT_ * D_, 256), blk, 0, stream>>>(
        convbuf, P(base + 13), NTOT_);
    k_logits_max<<<cdiv_ll((long long)E_ * H_, 256), blk, 0, stream>>>(
        asrc, adst, aeb, srcI, dstI, lgbuf, mxb);
    k_exp_den<<<cdiv_ll((long long)E_ * H_, 256), blk, 0, stream>>>(
        lgbuf, mxb, dstI, denb);
    k_scatter_msg<<<cdiv_ll((long long)E_ * H_, 256), blk, 0, stream>>>(
        lgbuf, denb, srcI, dstI, Hbuf, convbuf);
    // node = relu(LN(conv)) + node
    ln_relu_kernel<true><<<cdiv_ll(NTOT_, 8), blk, 0, stream>>>(
        convbuf, P(base + 21), P(base + 20), node, node, NTOT_);

    // ---- cross attention: cr = mha(node, memory, memory) + node ----
    gemm_wmma<0,false,false,true><<<ggrid(NTOT_), blk, 0, stream>>>(
        node, nullptr, nullptr, nullptr, nullptr, P(base + 2), P(base + 6), nullptr,
        qbuf, NTOT_, D_, D_);
    gemm_wmma<0,false,false,true><<<ggrid(B_ * S_), blk, 0, stream>>>(
        memory, nullptr, nullptr, nullptr, nullptr, P(base + 0), P(base + 4), nullptr,
        kbuf, B_ * S_, D_, D_);
    gemm_wmma<0,false,false,true><<<ggrid(B_ * S_), blk, 0, stream>>>(
        memory, nullptr, nullptr, nullptr, nullptr, P(base + 3), P(base + 7), nullptr,
        vbuf, B_ * S_, D_, D_);
    attn_kernel<<<dim3(B_, H_), blk, 0, stream>>>(qbuf, kbuf, vbuf, attnbuf,
                                                  M_, S_, scale, 0);
    gemm_wmma<0,false,true,true><<<ggrid(NTOT_), blk, 0, stream>>>(
        attnbuf, nullptr, nullptr, nullptr, nullptr, P(base + 1), P(base + 5), node,
        crbuf, NTOT_, D_, D_);
    // node = relu(LN(cr))
    ln_relu_kernel<false><<<cdiv_ll(NTOT_, 8), blk, 0, stream>>>(
        crbuf, P(base + 23), P(base + 22), nullptr, node, NTOT_);

    // ---- edge update ----
    // T1 = relu(concat(edge, node[src], node[dst]) @ W1 + b1)   (z never materialized)
    gemm_wmma<2,true,false,true><<<ggrid(E_), blk, 0, stream>>>(
        nullptr, edge, node, srcI, dstI, P(base + 14), P(base + 16), nullptr,
        Ebuf, E_, D_, 3 * D_);
    // edge <- T1 @ W2 + b2 + edge   (residual fused; element read-before-write)
    gemm_wmma<0,false,true,true><<<ggrid(E_), blk, 0, stream>>>(
        Ebuf, nullptr, nullptr, nullptr, nullptr, P(base + 15), P(base + 17), edge,
        edge, E_, D_, D_);
    // edge = relu(LN(edge))  in-place
    ln_relu_kernel<false><<<cdiv_ll(E_, 8), blk, 0, stream>>>(
        edge, P(base + 19), P(base + 18), nullptr, edge, E_);
  }
}
